// ScaledDotProductAttention_Operator_80350248173816
// MI455X (gfx1250) — compile-verified, matrix operand order checked by reading, not run on hardware
//
#include <hip/hip_runtime.h>
#include <hip/hip_bf16.h>
#include <math.h>

// Shapes: b=8, n=8, p=64, x=32, y=32, d=8  ->  F = x*y*d = 8192, SCALE = 1024
// query/key/value: (b,n,p,F) f32 row-major; mask: (b,64) bool (1 byte)
// out: (b,p,F,n) f32.
// Workspace layout (needs 5 MB):
//   [0 , 1MB)  attn  : f32 [bn][64][64]
//   [1MB, 5MB) part  : f32 [bn][4 ksplit][64][64] partial QK^T scores

#define B_   8
#define N_   8
#define P_   64
#define F_   8192
#define BN_  64
#define KSPLIT 4
#define KCHUNK (F_ / KSPLIT)   // 2048

typedef __attribute__((ext_vector_type(16))) __bf16 v16bf;
typedef __attribute__((ext_vector_type(8)))  float  v8f;
typedef __attribute__((ext_vector_type(2)))  float  v2f;

// 16-bit A/B fragment for V_WMMA_F32_16X16X32_BF16 (ISA §7.12.2):
// lanes 0-15 hold row M=lane, K = {kk..kk+7, kk+16..kk+23}
// lanes 16-31 hold row M=lane-16, K = {kk+8..kk+15, kk+24..kk+31}
__device__ __forceinline__ v16bf load_frag_bf16(const float* __restrict__ row,
                                                int kk, int half) {
  const float4* p0 = reinterpret_cast<const float4*>(row + kk + half * 8);
  float4 f0 = p0[0];
  float4 f1 = p0[1];
  const float4* p1 = reinterpret_cast<const float4*>(row + kk + 16 + half * 8);
  float4 f2 = p1[0];
  float4 f3 = p1[1];
  v16bf r;
  r[0]  = (__bf16)f0.x; r[1]  = (__bf16)f0.y; r[2]  = (__bf16)f0.z; r[3]  = (__bf16)f0.w;
  r[4]  = (__bf16)f1.x; r[5]  = (__bf16)f1.y; r[6]  = (__bf16)f1.z; r[7]  = (__bf16)f1.w;
  r[8]  = (__bf16)f2.x; r[9]  = (__bf16)f2.y; r[10] = (__bf16)f2.z; r[11] = (__bf16)f2.w;
  r[12] = (__bf16)f3.x; r[13] = (__bf16)f3.y; r[14] = (__bf16)f3.z; r[15] = (__bf16)f3.w;
  return r;
}

// ---------------------------------------------------------------------------
// Kernel 1: partial S = Q . K^T over a 2048-wide K slice, bf16 WMMA.
// grid.x = 64 bn * 4 ksplit; block = 256 (8 waves).
// wave w: row-tile m = w>>1 (16 rows), column pair c = w&1 (two 16-col tiles).
// Pure 256 MB stream -> prefetch 2 KB ahead on each of the three row streams.
// ---------------------------------------------------------------------------
__global__ __launch_bounds__(256) void qk_partial(const float* __restrict__ Q,
                                                  const float* __restrict__ K,
                                                  float* __restrict__ part) {
  const int blk  = blockIdx.x;
  const int bn   = blk >> 2;
  const int ks   = blk & 3;
  const int t    = threadIdx.x;
  const int w    = t >> 5;
  const int lane = t & 31;
  const int half = lane >> 4;
  const int lrow = lane & 15;
  const int m    = w >> 1;   // 0..3
  const int c    = w & 1;    // 0..1

  const float* qrow  = Q + ((size_t)bn * P_ + (size_t)(m * 16 + lrow)) * F_;
  const float* krow0 = K + ((size_t)bn * P_ + (size_t)(c * 32 + lrow)) * F_;
  const float* krow1 = krow0 + (size_t)16 * F_;

  v8f c0 = {};
  v8f c1 = {};
  const int kbase = ks * KCHUNK;
  for (int kk = kbase; kk < kbase + KCHUNK; kk += 32) {
    __builtin_prefetch(qrow  + kk + 512, 0, 1);   // global_prefetch_b8, 2KB ahead
    __builtin_prefetch(krow0 + kk + 512, 0, 1);
    __builtin_prefetch(krow1 + kk + 512, 0, 1);
    v16bf a  = load_frag_bf16(qrow,  kk, half);
    v16bf b0 = load_frag_bf16(krow0, kk, half);
    v16bf b1 = load_frag_bf16(krow1, kk, half);
    c0 = __builtin_amdgcn_wmma_f32_16x16x32_bf16(false, a, false, b0,
                                                 (short)0, c0, false, false);
    c1 = __builtin_amdgcn_wmma_f32_16x16x32_bf16(false, a, false, b1,
                                                 (short)0, c1, false, false);
  }

  // C/D layout: lanes 0-15: VGPR r -> (row m*16+r, col lane);
  //             lanes 16-31: VGPR r -> (row m*16+8+r, col lane-16)
  float* dst = part + (size_t)blk * (P_ * P_);
  const int row0 = m * 16 + half * 8;
  const int col0 = c * 32 + lrow;
#pragma unroll
  for (int r = 0; r < 8; r++) {
    dst[(row0 + r) * P_ + col0]      = c0[r];
    dst[(row0 + r) * P_ + col0 + 16] = c1[r];
  }
}

// ---------------------------------------------------------------------------
// Kernel 2: reduce 4 partials, scale, mask, exact fp32 softmax -> attn.
// grid.x = 64 (bn); block = 64 threads, one row p each.
// ---------------------------------------------------------------------------
__global__ __launch_bounds__(64) void softmax_kernel(const float* __restrict__ part,
                                                     const unsigned char* __restrict__ mask,
                                                     float* __restrict__ attn) {
  const int bn = blockIdx.x;
  const int p  = threadIdx.x;
  const int b  = bn >> 3;
  const float* base = part + (size_t)bn * KSPLIT * (P_ * P_) + (size_t)p * P_;

  float s[64];
  float mx = -INFINITY;
#pragma unroll
  for (int q = 0; q < 64; q++) {
    float v = base[q] + base[P_ * P_ + q] + base[2 * P_ * P_ + q] + base[3 * P_ * P_ + q];
    v *= (1.0f / 1024.0f);
    if (mask[b * P_ + q]) v = -INFINITY;
    s[q] = v;
    mx = fmaxf(mx, v);
  }
  float sum = 0.0f;
#pragma unroll
  for (int q = 0; q < 64; q++) {
    float e = __expf(s[q] - mx);
    s[q] = e;
    sum += e;
  }
  const float inv = 1.0f / sum;
  float* dst = attn + (size_t)bn * (P_ * P_) + (size_t)p * P_;
#pragma unroll
  for (int q = 0; q < 64; q++) dst[q] = s[q] * inv;
}

// ---------------------------------------------------------------------------
// Kernel 3: O[b][p][f][n] = sum_q attn[b][n][p][q] * V[b][n][q][f]
// Exact fp32 WMMA (16x16x4). attn[b] (128 KB) staged to LDS with the CDNA5
// async pipe (GLOBAL_LOAD_ASYNC_TO_LDS_B128, tracked by ASYNCcnt) so the
// stage never touches VGPR return paths; padded row stride 68 kills LDS bank
// conflicts for the per-lane ds_load_b64 A-fragments. All 8 n's accumulate in
// registers so the n-interleaved output transpose is two contiguous float4
// stores per element group.
// grid = (64 f-chunks of 128, 8 b); block = 256 (8 waves, wave w owns 16 f's).
// ---------------------------------------------------------------------------
#define APAD 68   // padded attn row stride in floats (16B-aligned, conflict-free)

__global__ __launch_bounds__(256) void av_out(const float* __restrict__ V,
                                              const float* __restrict__ attn,
                                              float* __restrict__ out) {
  __shared__ float at[N_ * P_ * APAD];   // 8*64*68*4 = 136 KB of the 320 KB LDS

  const int fc = blockIdx.x;
  const int b  = blockIdx.y;
  const int t  = threadIdx.x;

  // Async-stage attn[b] (8 x 64 x 64 f32) into padded LDS.
  // 8192 float4 chunks; chunk e -> row r = e>>4, 16B column (e&15).
  {
    const float* gbase = attn + (size_t)b * (N_ * P_ * P_);
#pragma unroll
    for (int i = 0; i < 32; i++) {
      const int e   = t + i * 256;
      const int r   = e >> 4;
      const unsigned goff = (unsigned)e * 16u;                       // bytes
      const unsigned loff = (unsigned)(r * APAD + (e & 15) * 4) * 4u; // bytes
      unsigned lds_vaddr = (unsigned)(size_t)(at) + loff;            // LDS byte addr
      asm volatile("global_load_async_to_lds_b128 %0, %1, %2"
                   :: "v"(lds_vaddr), "v"(goff), "s"(gbase)
                   : "memory");
    }
    asm volatile("s_wait_asynccnt 0x0" ::: "memory");
  }
  __syncthreads();

  const int w    = t >> 5;
  const int lane = t & 31;
  const int half = lane >> 4;
  const int lrow = lane & 15;
  const int fcol = fc * 128 + w * 16 + lrow;

  for (int pb = 0; pb < 4; pb++) {
    float nbuf[8][8];
    const int prow = pb * 16 + lrow;   // A-matrix row M = lrow for both halves
#pragma unroll
    for (int n = 0; n < N_; n++) {
      const float* Vn = V + ((size_t)(b * N_ + n) * P_) * F_;
      const float* an = at + (n * P_ + prow) * APAD;
      v8f acc = {};
#pragma unroll
      for (int j = 0; j < 16; j++) {
        // f32 16x16x4 layout: lanes 0-15 hold K = 4j,4j+1 ; lanes 16-31 K = 4j+2,4j+3
        const int q = 4 * j + half * 2;
        float2 af = *reinterpret_cast<const float2*>(an + q);
        v2f a;  a[0] = af.x; a[1] = af.y;
        v2f bb; bb[0] = Vn[(size_t)q * F_ + fcol];
                bb[1] = Vn[(size_t)(q + 1) * F_ + fcol];
        acc = __builtin_amdgcn_wmma_f32_16x16x4_f32(false, a, false, bb,
                                                    (short)0, acc, false, false);
      }
#pragma unroll
      for (int r = 0; r < 8; r++) nbuf[r][n] = acc[r];
    }
    // Lane owns (p = pb*16 + half*8 + r, f = fcol); n runs contiguous in out.
#pragma unroll
    for (int r = 0; r < 8; r++) {
      const int p = pb * 16 + half * 8 + r;
      float* o = out + (((size_t)(b * P_ + p)) * F_ + (size_t)fcol) * N_;
      reinterpret_cast<float4*>(o)[0] =
          make_float4(nbuf[r][0], nbuf[r][1], nbuf[r][2], nbuf[r][3]);
      reinterpret_cast<float4*>(o)[1] =
          make_float4(nbuf[r][4], nbuf[r][5], nbuf[r][6], nbuf[r][7]);
    }
  }
}

// ---------------------------------------------------------------------------
extern "C" void kernel_launch(void* const* d_in, const int* in_sizes, int n_in,
                              void* d_out, int out_size, void* d_ws, size_t ws_size,
                              hipStream_t stream) {
  const float* Q = (const float*)d_in[0];
  const float* K = (const float*)d_in[1];
  const float* V = (const float*)d_in[2];
  const unsigned char* mask = (const unsigned char*)d_in[3];
  float* out = (float*)d_out;

  float* attn = (float*)d_ws;                       // 64*64*64 f32 = 1 MB
  float* part = attn + (size_t)BN_ * P_ * P_;       // 64*4*64*64 f32 = 4 MB

  qk_partial<<<dim3(BN_ * KSPLIT), dim3(256), 0, stream>>>(Q, K, part);
  softmax_kernel<<<dim3(BN_), dim3(64), 0, stream>>>(part, mask, attn);
  av_out<<<dim3(F_ / 128, B_), dim3(256), 0, stream>>>(V, attn, out);
}
